// SALSA_30485677867538
// MI455X (gfx1250) — matrix operand order repaired and checked by reading, not compiled, not run on hardware
//
#include <hip/hip_runtime.h>
#include <hip/hip_bf16.h>
#include <math.h>

// ---------------- problem constants (fixed by the reference) ----------------
#define BATCH 4
#define CIN   64
#define ICH   16
#define HH    256
#define WWI   256
#define PIX   (HH * WWI)        // 65536
#define LDIM  64                // Lh == Lw
#define NP    (LDIM * LDIM)     // 4096 patches
#define DREAL 784               // 16 * 7 * 7
#define DP    800               // padded to multiple of 32
#define PPITCH 8192             // f16 pitch of P rows, aliased over S (byte pitch 16384)
#define SCALE 10.0f

#if defined(__gfx1250__) && __has_builtin(__builtin_amdgcn_global_load_async_to_lds_b128) && \
    __has_builtin(__builtin_amdgcn_s_wait_asynccnt)
#define HAVE_ASYNC 1
#else
#define HAVE_ASYNC 0
#endif

typedef __attribute__((ext_vector_type(16))) _Float16 v16h;
typedef __attribute__((ext_vector_type(8)))  _Float16 v8h;
typedef __attribute__((ext_vector_type(8)))  float    v8f;
typedef __attribute__((ext_vector_type(4)))  int      v4i;

#if HAVE_ASYNC
typedef __attribute__((address_space(1))) v4i* as1_v4i_ptr;
typedef __attribute__((address_space(3))) v4i* as3_v4i_ptr;
#endif

static __device__ __forceinline__ int imin(int a, int b) { return a < b ? a : b; }
static __device__ __forceinline__ int imax(int a, int b) { return a > b ? a : b; }

// ---- WMMA fragment loads (work for global or LDS pointers; pitch in halves) ----
// A-matrix 16x32 f16: lane L: M=L&15, half=L>>4; VGPR0-3: K=half*8+0..7, VGPR4-7: K=16+half*8+0..7
static __device__ __forceinline__ v16h frag_a(const _Float16* base, int pitch, int r0, int k0) {
  int lane = threadIdx.x & 31;
  int m = lane & 15, h8 = (lane >> 4) << 3;
  const _Float16* p = base + (size_t)(r0 + m) * (size_t)pitch + k0;
  v8h lo = *(const v8h*)(p + h8);
  v8h hi = *(const v8h*)(p + 16 + h8);
  v16h out;
#pragma unroll
  for (int i = 0; i < 8; ++i) { out[i] = lo[i]; out[i + 8] = hi[i]; }
  return out;
}

// B-matrix 32x16 f16, source is B^T rows (Bsrc[n][k]): lane L: N=L&15, half=L>>4; elems 0-7: K=half*8+0..7, elems 8-15: K=16+half*8+0..7
static __device__ __forceinline__ v16h frag_b(const _Float16* base, int pitch, int n0, int k0) {
  int lane = threadIdx.x & 31;
  int n = lane & 15, h8 = (lane >> 4) << 3;
  const _Float16* p = base + (size_t)(n0 + n) * (size_t)pitch + k0;
  v8h lo = *(const v8h*)(p + h8);
  v8h hi = *(const v8h*)(p + 16 + h8);
  v16h out;
#pragma unroll
  for (int i = 0; i < 8; ++i) { out[i] = lo[i]; out[i + 8] = hi[i]; }
  return out;
}

// ---- tile loader: global (row-major, 32-half K slab, 64B rows) -> LDS [rows][32] ----
// N128 = number of 16-byte transfers = rows*4. 256-thread blocks.
template <int N128>
static __device__ __forceinline__ void tile_load(_Float16* lds, const _Float16* g,
                                                 int grow0, int gcol0, int gpitch) {
  for (int idx = threadIdx.x; idx < N128; idx += 256) {
    int row = idx >> 2;            // 4 x b128 per 64-byte row
    int col = (idx & 3) << 3;      // halves
    const _Float16* src = g + (size_t)(grow0 + row) * (size_t)gpitch + gcol0 + col;
    _Float16* dst = lds + row * 32 + col;
#if HAVE_ASYNC
    __builtin_amdgcn_global_load_async_to_lds_b128(
        (as1_v4i_ptr)src, (as3_v4i_ptr)dst, 0, 0);
#else
    *(v8h*)dst = *(const v8h*)src;
#endif
  }
}

static __device__ __forceinline__ void waitbar() {
#if HAVE_ASYNC
  __builtin_amdgcn_s_wait_asynccnt(0);
#endif
  __syncthreads();
}

// ---------------- kernel 1: fused 1x1 convs -> f16 q/k/v maps ----------------
// q <- (Wg,bg), k <- (Wp,bp), v <- (Wt,bt)   [reference mapping]
__global__ void conv_qkv(const float* __restrict__ x,
                         const float* __restrict__ Wg, const float* __restrict__ bg,
                         const float* __restrict__ Wt, const float* __restrict__ bt,
                         const float* __restrict__ Wp, const float* __restrict__ bp,
                         _Float16* __restrict__ qm, _Float16* __restrict__ km,
                         _Float16* __restrict__ vm) {
  int pix = blockIdx.x * blockDim.x + threadIdx.x;
  if (pix >= PIX) return;
  float xi[CIN];
#pragma unroll
  for (int c = 0; c < CIN; ++c) xi[c] = x[(size_t)c * PIX + pix];
  for (int o = 0; o < ICH; ++o) {
    float aq = bg[o], ak = bp[o], av = bt[o];
#pragma unroll
    for (int c = 0; c < CIN; ++c) {
      float xv = xi[c];
      aq = fmaf(Wg[o * CIN + c], xv, aq);
      ak = fmaf(Wp[o * CIN + c], xv, ak);
      av = fmaf(Wt[o * CIN + c], xv, av);
    }
    qm[(size_t)o * PIX + pix] = (_Float16)aq;
    km[(size_t)o * PIX + pix] = (_Float16)ak;
    vm[(size_t)o * PIX + pix] = (_Float16)av;
  }
}

// ---------------- kernel 2: unfold into padded patch matrices ----------------
// SAME pad for H=256,k=7,s=4: pad=(1,2). Q,K: [NP][DP]; V stored transposed: Vt[DP][NP].
__global__ void build_patches(const _Float16* __restrict__ qm, const _Float16* __restrict__ km,
                              const _Float16* __restrict__ vm,
                              _Float16* __restrict__ Q, _Float16* __restrict__ Kp,
                              _Float16* __restrict__ Vt) {
  int idx = blockIdx.x * blockDim.x + threadIdx.x;
  if (idx >= NP * DP) return;
  int d = idx % DP;
  int p = idx / DP;
  _Float16 q = (_Float16)0.f, k = (_Float16)0.f, v = (_Float16)0.f;
  if (d < DREAL) {
    int c  = d / 49;
    int rr = (d % 49) / 7;
    int cc = d % 7;
    int pi = p >> 6, pj = p & 63;
    int y = pi * 4 + rr - 1;      // pad top = 1
    int xw = pj * 4 + cc - 1;
    if (y >= 0 && y < HH && xw >= 0 && xw < WWI) {
      size_t off = (size_t)c * PIX + (size_t)y * WWI + xw;
      q = qm[off]; k = km[off]; v = vm[off];
    }
  }
  Q [(size_t)p * DP + d] = q;
  Kp[(size_t)p * DP + d] = k;
  Vt[(size_t)d * NP + p] = v;
}

// ---------------- kernel 3: S = (Q @ K^T) * SCALE via WMMA ----------------
// 256 threads = 8 waves; block tile 128x128; LDS double-buffered 32-wide K slabs.
__global__ void __launch_bounds__(256) gemm_qk(const _Float16* __restrict__ Q,
                                               const _Float16* __restrict__ Kp,
                                               float* __restrict__ S) {
  __shared__ _Float16 As[2][128 * 32];
  __shared__ _Float16 Bs[2][128 * 32];
  int bm = blockIdx.x >> 5;            // 0..31
  int bn = blockIdx.x & 31;            // 0..31
  int m0 = bm * 128, n0 = bn * 128;
  int wave = threadIdx.x >> 5;

  v8f acc[8] = {};
  tile_load<512>(As[0], Q,  m0, 0, DP);
  tile_load<512>(Bs[0], Kp, n0, 0, DP);
  waitbar();

  for (int kk = 0; kk < DP / 32; ++kk) {
    int cur = kk & 1, nxt = cur ^ 1;
    if (kk + 1 < DP / 32) {
      tile_load<512>(As[nxt], Q,  m0, (kk + 1) * 32, DP);
      tile_load<512>(Bs[nxt], Kp, n0, (kk + 1) * 32, DP);
    }
    v16h a = frag_a(As[cur], 32, wave * 16, 0);
#pragma unroll
    for (int j = 0; j < 8; ++j) {
      v16h b = frag_b(Bs[cur], 32, j * 16, 0);
      acc[j] = __builtin_amdgcn_wmma_f32_16x16x32_f16(false, a, false, b,
                                                      (short)0, acc[j], false, false);
    }
    waitbar();
  }

  int lane = threadIdx.x & 31;
  int cn = lane & 15, rb = (lane >> 4) << 3;
#pragma unroll
  for (int j = 0; j < 8; ++j)
#pragma unroll
    for (int r = 0; r < 8; ++r)
      S[(size_t)(m0 + wave * 16 + rb + r) * NP + (n0 + j * 16 + cn)] = acc[j][r] * SCALE;
}

// ---------------- kernel 4: row softmax, write f16 P in place over S ----------------
__global__ void softmax_rows(float* __restrict__ Sbuf) {
  __shared__ float buf[NP];
  __shared__ float red[256];
  int tid = threadIdx.x;
  size_t row = blockIdx.x;
  float* srow = Sbuf + row * NP;
  float mx = -3.4e38f;
  for (int i = tid; i < NP; i += 256) { float v = srow[i]; buf[i] = v; mx = fmaxf(mx, v); }
  red[tid] = mx; __syncthreads();
  for (int s = 128; s > 0; s >>= 1) {
    if (tid < s) red[tid] = fmaxf(red[tid], red[tid + s]);
    __syncthreads();
  }
  float m = red[0]; __syncthreads();
  float sm = 0.f;
  for (int i = tid; i < NP; i += 256) { float e = __expf(buf[i] - m); buf[i] = e; sm += e; }
  red[tid] = sm; __syncthreads();
  for (int s = 128; s > 0; s >>= 1) {
    if (tid < s) red[tid] += red[tid + s];
    __syncthreads();
  }
  float inv = 1.0f / red[0];
  _Float16* prow = (_Float16*)Sbuf + row * PPITCH;   // same bytes as this row's S
  for (int i = tid; i < NP; i += 256) prow[i] = (_Float16)(buf[i] * inv);
}

// ---------------- kernel 5: O = P @ V via WMMA (B from V^T) ----------------
// 256 threads = 8 waves; block tile 128x160 (160 divides DP=800); K = NP in 128 steps.
__global__ void __launch_bounds__(256) gemm_pv(const _Float16* __restrict__ P,
                                               const _Float16* __restrict__ Vt,
                                               float* __restrict__ O) {
  __shared__ _Float16 As[2][128 * 32];
  __shared__ _Float16 Bs[2][160 * 32];
  int bm = blockIdx.x / 5;             // 0..31
  int bn = blockIdx.x % 5;             // 0..4
  int m0 = bm * 128, n0 = bn * 160;
  int wave = threadIdx.x >> 5;

  v8f acc[10] = {};
  tile_load<512>(As[0], P,  m0, 0, PPITCH);
  tile_load<640>(Bs[0], Vt, n0, 0, NP);
  waitbar();

  for (int kk = 0; kk < NP / 32; ++kk) {
    int cur = kk & 1, nxt = cur ^ 1;
    if (kk + 1 < NP / 32) {
      tile_load<512>(As[nxt], P,  m0, (kk + 1) * 32, PPITCH);
      tile_load<640>(Bs[nxt], Vt, n0, (kk + 1) * 32, NP);
    }
    v16h a = frag_a(As[cur], 32, wave * 16, 0);
#pragma unroll
    for (int j = 0; j < 10; ++j) {
      v16h b = frag_b(Bs[cur], 32, j * 16, 0);
      acc[j] = __builtin_amdgcn_wmma_f32_16x16x32_f16(false, a, false, b,
                                                      (short)0, acc[j], false, false);
    }
    waitbar();
  }

  int lane = threadIdx.x & 31;
  int cn = lane & 15, rb = (lane >> 4) << 3;
#pragma unroll
  for (int j = 0; j < 10; ++j)
#pragma unroll
    for (int r = 0; r < 8; ++r)
      O[(size_t)(m0 + wave * 16 + rb + r) * DP + (n0 + j * 16 + cn)] = acc[j][r];
}

// ---------------- kernel 6: fold (overlap-add) + divide by coverage mask ----------------
__global__ void fold_zi(const float* __restrict__ O, float* __restrict__ Z) {
  int idx = blockIdx.x * blockDim.x + threadIdx.x;
  if (idx >= ICH * PIX) return;
  int pix = idx & (PIX - 1);
  int c   = idx >> 16;
  int h = pix >> 8, w = pix & 255;
  int hp = h + 3, wp = w + 3;
  int i0 = imax(0, (hp - 6 + 3) >> 2), i1 = imin(LDIM - 1, hp >> 2);
  int j0 = imax(0, (wp - 6 + 3) >> 2), j1 = imin(LDIM - 1, wp >> 2);
  float sum = 0.f;
  int cnt = 0;
  for (int i = i0; i <= i1; ++i) {
    int kh = hp - 4 * i;
    if (kh < 0 || kh > 6) continue;
    for (int j = j0; j <= j1; ++j) {
      int kw = wp - 4 * j;
      if (kw < 0 || kw > 6) continue;
      sum += O[(size_t)(i * LDIM + j) * DP + (c * 49 + kh * 7 + kw)];
      ++cnt;
    }
  }
  Z[idx] = sum / (float)cnt;
}

// ---------------- kernel 7: y = x + Ww @ (zi/mask) + bw ----------------
__global__ void final_out(const float* __restrict__ x, const float* __restrict__ Z,
                          const float* __restrict__ Ww, const float* __restrict__ bw,
                          float* __restrict__ y) {
  int idx = blockIdx.x * blockDim.x + threadIdx.x;
  if (idx >= CIN * PIX) return;
  int pix = idx & (PIX - 1);
  int c   = idx >> 16;
  float acc = bw[c];
#pragma unroll
  for (int ic = 0; ic < ICH; ++ic)
    acc = fmaf(Ww[c * ICH + ic], Z[(size_t)ic * PIX + pix], acc);
  y[idx] = x[idx] + acc;
}

// ---------------- host-side orchestration ----------------
extern "C" void kernel_launch(void* const* d_in, const int* in_sizes, int n_in,
                              void* d_out, int out_size, void* d_ws, size_t ws_size,
                              hipStream_t stream) {
  const float* xb = (const float*)d_in[0];
  const float* Wg = (const float*)d_in[1];
  const float* bg = (const float*)d_in[2];
  const float* Wt = (const float*)d_in[3];
  const float* bt = (const float*)d_in[4];
  const float* Wp = (const float*)d_in[5];
  const float* bp = (const float*)d_in[6];
  const float* Ww = (const float*)d_in[7];
  const float* bw = (const float*)d_in[8];
  float* y = (float*)d_out;

  char* ws = (char*)d_ws;
  const size_t SZ_S  = (size_t)NP * NP * sizeof(float);      // 64 MiB, also hosts qkv maps & P
  const size_t SZ_QK = (size_t)NP * DP * sizeof(_Float16);   // 6.25 MiB
  const size_t SZ_O  = (size_t)NP * DP * sizeof(float);
  float*    Sb = (float*)(ws);
  _Float16* qm = (_Float16*)(ws);                            // alias: used before S is written
  _Float16* km = (_Float16*)(ws + (size_t)ICH * PIX * 2);
  _Float16* vm = (_Float16*)(ws + (size_t)ICH * PIX * 4);
  _Float16* Q  = (_Float16*)(ws + SZ_S);
  _Float16* Kp = (_Float16*)(ws + SZ_S + SZ_QK);
  _Float16* Vt = (_Float16*)(ws + SZ_S + 2 * SZ_QK);
  float*    O  = (float*)   (ws + SZ_S + 3 * SZ_QK);
  float*    Z  = (float*)   (ws + SZ_S + 3 * SZ_QK + SZ_O);

  for (int b = 0; b < BATCH; ++b) {
    const float* x = xb + (size_t)b * CIN * PIX;
    float*       o = y  + (size_t)b * CIN * PIX;

    conv_qkv     <<<PIX / 256, 256, 0, stream>>>(x, Wg, bg, Wt, bt, Wp, bp, qm, km, vm);
    build_patches<<<(NP * DP) / 256, 256, 0, stream>>>(qm, km, vm, Q, Kp, Vt);
    gemm_qk      <<<32 * 32, 256, 0, stream>>>(Q, Kp, Sb);
    softmax_rows <<<NP, 256, 0, stream>>>(Sb);
    gemm_pv      <<<32 * 5, 256, 0, stream>>>((const _Float16*)Sb, Vt, O);
    fold_zi      <<<(ICH * PIX) / 256, 256, 0, stream>>>(O, Z);
    final_out    <<<(CIN * PIX) / 256, 256, 0, stream>>>(x, Z, Ww, bw, o);
  }
}
